// AnswerFinder_85933705659094
// MI455X (gfx1250) — matrix operand order repaired and checked by reading, not compiled, not run on hardware
//
#include <hip/hip_runtime.h>

typedef _Float16 h4   __attribute__((ext_vector_type(4)));
typedef _Float16 h8   __attribute__((ext_vector_type(8)));
typedef _Float16 v16h __attribute__((ext_vector_type(16)));
typedef float    v8f  __attribute__((ext_vector_type(8)));
typedef float    v4f  __attribute__((ext_vector_type(4)));
typedef unsigned int u32x4 __attribute__((ext_vector_type(4)));
typedef int          i32x8 __attribute__((ext_vector_type(8)));
typedef int          i32x4 __attribute__((ext_vector_type(4)));

#define BM 128
#define BN 64
#define BK 32

// gelu(tanh approx): 0.5x(1+tanh(z)) == x*sigmoid(2z), branch-free via native exp
__device__ __forceinline__ float gelu_tanh(float x) {
    float u = -1.5957691216057308f * (x + 0.044715f * x * x * x);
    return x / (1.0f + __expf(u));
}

__device__ __forceinline__ float reduce16(float v) {
#pragma unroll
    for (int off = 8; off > 0; off >>= 1)
        v += __shfl_xor(v, off, 16);
    return v;
}

// ---- Tensor Data Mover: 2D f16 tile load, D# per CDNA5 ISA ch.8 ----
// group0: count=1 | lds_addr | global_addr[31:0] | global_addr[56:32] + type=2
// group1: data_size=2B; tensor dims == tile dims (tile always in-bounds);
//         dim0 stride = row stride in elements.
__device__ __forceinline__ void tdm_load_2d(unsigned lds_off, const _Float16* gptr,
                                            unsigned tile_w, unsigned tile_h,
                                            unsigned stride_elems) {
    unsigned long long ga = (unsigned long long)(uintptr_t)gptr;
    u32x4 g0;
    g0[0] = 1u;                                              // count=1 (valid), user mode
    g0[1] = lds_off;                                         // LDS byte address
    g0[2] = (unsigned)(ga & 0xFFFFFFFFull);                  // global_addr[31:0]
    g0[3] = ((unsigned)(ga >> 32) & 0x01FFFFFFu) | 0x80000000u;  // [56:32] | type=2
    i32x8 g1;
    g1[0] = 0x00010000;                  // workgroup_mask=0, data_size=1 (2 bytes)
    g1[1] = (int)(tile_w << 16);         // tensor_dim0[15:0]
    g1[2] = (int)(tile_h << 16);         // tensor_dim1[15:0] (dim0[31:16]=0)
    g1[3] = (int)(tile_w << 16);         // tile_dim0 (dim1[31:16]=0)
    g1[4] = (int)tile_h;                 // tile_dim1 (tile_dim2=0)
    g1[5] = (int)stride_elems;           // tensor_dim0_stride[31:0]
    g1[6] = 0;                           // stride0[47:32]=0, stride1[15:0]=0
    g1[7] = 0;                           // stride1[47:16]=0
    i32x4 z4 = {0, 0, 0, 0};             // groups 2/3 unused (2D tensor)
    i32x8 z8 = {0, 0, 0, 0, 0, 0, 0, 0};
    __builtin_amdgcn_tensor_load_to_lds(g0, g1, z4, z4, z8, 0);  // clang-23 6-arg form
}

// MODE 0: store H (f16), fused dots dvecA->dotA, dvecB->dotB
// MODE 1: no store, fused dot dvecB->dotB only (cond never materialized)
template <int MODE>
__global__ __launch_bounds__(256) void fused_gemm_tdm_kernel(
    const _Float16* __restrict__ A16, const _Float16* __restrict__ WT16,
    const float* __restrict__ bias, const unsigned char* __restrict__ mask,
    const float* __restrict__ dvecA, const float* __restrict__ dvecB,
    _Float16* __restrict__ Hout, float* __restrict__ dotA, float* __restrict__ dotB,
    int M, int N, int K) {
    __shared__ alignas(16) _Float16 As[2][BM * BK];   // [row][k] f16, 8 KB per buffer
    __shared__ alignas(16) _Float16 Bs[2][BN * BK];   // [col][k] f16 (from W^T), 4 KB

    const int tid  = threadIdx.x;
    const int lane = tid & 31;
    const int wave = tid >> 5;
    const int wm   = wave >> 1;          // 0..3 -> 32 rows each
    const int wn   = wave & 1;           // 0..1 -> 32 cols each
    const int bm   = blockIdx.y * BM;
    const int bn   = blockIdx.x * BN;
    const int half = lane >> 4;
    const int ln   = lane & 15;

    const unsigned as_off[2] = { (unsigned)(uintptr_t)&As[0][0], (unsigned)(uintptr_t)&As[1][0] };
    const unsigned bs_off[2] = { (unsigned)(uintptr_t)&Bs[0][0], (unsigned)(uintptr_t)&Bs[1][0] };
    const _Float16* Abase = A16  + (size_t)bm * K;
    const _Float16* Wbase = WT16 + (size_t)bn * K;

    v8f acc[2][2] = {};

    const int nk = K / BK;
    if (wave == 0) {                      // prologue: fill buffer 0
        tdm_load_2d(as_off[0], Abase, BK, BM, K);
        tdm_load_2d(bs_off[0], Wbase, BK, BN, K);
    }
    for (int it = 0; it < nk; ++it) {
        const int buf = it & 1;
        if (wave == 0) {
            if (it + 1 < nk) {            // prefetch next tile pair into other buffer
                tdm_load_2d(as_off[buf ^ 1], Abase + (it + 1) * BK, BK, BM, K);
                tdm_load_2d(bs_off[buf ^ 1], Wbase + (it + 1) * BK, BK, BN, K);
                __builtin_amdgcn_s_wait_tensorcnt(2);   // current pair complete (in-order)
            } else {
                __builtin_amdgcn_s_wait_tensorcnt(0);
            }
        }
        __syncthreads();                  // data ready for all waves

        const _Float16* asb = As[buf];
        const _Float16* bsb = Bs[buf];
        // A fragment: lanes 0-15: K {0..15}; lanes 16-31: K {16..31}
        v16h afr[2];
#pragma unroll
        for (int mi = 0; mi < 2; ++mi) {
            int row = wm * 32 + mi * 16 + ln;
            h8 lo = *(const h8*)&asb[row * BK + half * 16];
            h8 hi = *(const h8*)&asb[row * BK + half * 16 + 8];
#pragma unroll
            for (int j = 0; j < 8; ++j) { afr[mi][j] = lo[j]; afr[mi][8 + j] = hi[j]; }
        }
        // B fragment: lane = column, K contiguous (0..15 / 16..31 per half-wave)
        v16h bfr[2];
#pragma unroll
        for (int ni = 0; ni < 2; ++ni) {
            int col = wn * 32 + ni * 16 + ln;
            bfr[ni] = *(const v16h*)&bsb[col * BK + half * 16];
        }
#pragma unroll
        for (int mi = 0; mi < 2; ++mi)
#pragma unroll
            for (int ni = 0; ni < 2; ++ni)
                acc[mi][ni] = __builtin_amdgcn_wmma_f32_16x16x32_f16(
                    false, afr[mi], false, bfr[ni], (short)0, acc[mi][ni], false, false);

        __syncthreads();                  // reads done before this buffer is overwritten
    }

    // ---- epilogue: bias + gelu + mask, optional f16 store, fused row-dots ----
    int   coln[2];
    float bcol[2], dAc[2] = {0.f, 0.f}, dBc[2];
#pragma unroll
    for (int ni = 0; ni < 2; ++ni) {
        coln[ni] = bn + wn * 32 + ni * 16 + ln;
        bcol[ni] = bias[coln[ni]];
        dBc[ni]  = dvecB[coln[ni]];
        if (MODE == 0) dAc[ni] = dvecA[coln[ni]];
    }
#pragma unroll
    for (int mi = 0; mi < 2; ++mi) {
#pragma unroll
        for (int r = 0; r < 8; ++r) {
            int grow   = bm + wm * 32 + mi * 16 + half * 8 + r;
            float mrow = mask[grow] ? 1.0f : 0.0f;
            float hv[2];
#pragma unroll
            for (int ni = 0; ni < 2; ++ni) {
                float x = acc[mi][ni][r] + bcol[ni];
                hv[ni]  = gelu_tanh(x) * mrow;
            }
            if (MODE == 0) {
                Hout[(size_t)grow * N + coln[0]] = (_Float16)hv[0];
                Hout[(size_t)grow * N + coln[1]] = (_Float16)hv[1];
                float p1 = reduce16(hv[0] * dAc[0] + hv[1] * dAc[1]);
                if (ln == 0) atomicAdd(&dotA[grow], p1);
            }
            float p2 = reduce16(hv[0] * dBc[0] + hv[1] * dBc[1]);
            if (ln == 0) atomicAdd(&dotB[grow], p2);
        }
    }
}

__global__ void zero_kernel(float* __restrict__ p, int n) {
    int i = blockIdx.x * blockDim.x + threadIdx.x;
    if (i < n) p[i] = 0.0f;
}

// f32 -> f16 elementwise (vector-4)
__global__ __launch_bounds__(256) void cvt_kernel(const float* __restrict__ src,
                                                  _Float16* __restrict__ dst, int n4) {
    int i = blockIdx.x * blockDim.x + threadIdx.x;
    if (i < n4) {
        v4f v = ((const v4f*)src)[i];
        h4 h;
#pragma unroll
        for (int j = 0; j < 4; ++j) h[j] = (_Float16)v[j];
        ((h4*)dst)[i] = h;
    }
}

// W [K][N] f32 -> WT [N][K] f16, 64x64 LDS tile transpose
__global__ __launch_bounds__(256) void transpose_cvt_kernel(
    const float* __restrict__ W, _Float16* __restrict__ WT, int Kd, int Nd) {
    __shared__ _Float16 t[64][64 + 8];
    const int k0 = blockIdx.y * 64, n0 = blockIdx.x * 64;
    const int c  = threadIdx.x & 63;
    const int r0 = threadIdx.x >> 6;      // 0..3
#pragma unroll
    for (int r = r0; r < 64; r += 4)
        t[c][r] = (_Float16)W[(size_t)(k0 + r) * Nd + n0 + c];   // t[n][k]
    __syncthreads();
#pragma unroll
    for (int r = r0; r < 64; r += 4)
        WT[(size_t)(n0 + r) * Kd + k0 + c] = t[r][c];
}

// Per-batch: start-logits LSE; end-logits LSE via suffix scans over e_i/e_j.
__global__ __launch_bounds__(1024) void batch_stats_kernel(
    const float* __restrict__ s_logit, const float* __restrict__ e_i,
    const float* __restrict__ e_j, const unsigned char* __restrict__ mask,
    float* __restrict__ slp, float* __restrict__ lse_e) {
    const int S = 1024;
    __shared__ float ls[1024];
    __shared__ float red[1024];
    const int b = blockIdx.x, t = threadIdx.x;
    const int idx = b * S + t;
    const bool mk = mask[idx] != 0;
    const float slv = mk ? s_logit[idx] : -10.0f;
    const float eiv = e_i[idx];
    const float ejv = e_j[idx];

    red[t] = slv; __syncthreads();
    for (int s = 512; s > 0; s >>= 1) { if (t < s) red[t] = fmaxf(red[t], red[t + s]); __syncthreads(); }
    float Ms = red[0]; __syncthreads();
    red[t] = __expf(slv - Ms); __syncthreads();
    for (int s = 512; s > 0; s >>= 1) { if (t < s) red[t] += red[t + s]; __syncthreads(); }
    float Zs = red[0]; __syncthreads();
    slp[idx] = (Ms + __logf(Zs)) - slv;

    ls[t] = mk ? ejv : -INFINITY; __syncthreads();
    for (int s = 1; s < S; s <<= 1) {
        float v = ls[t];
        float o = (t + s < S) ? ls[t + s] : -INFINITY;
        __syncthreads(); ls[t] = fmaxf(v, o); __syncthreads();
    }
    float sufmax = ls[t];
    red[t] = mk ? (eiv + sufmax) : -INFINITY; __syncthreads();
    for (int s = 512; s > 0; s >>= 1) { if (t < s) red[t] = fmaxf(red[t], red[t + s]); __syncthreads(); }
    float Mv = red[0]; __syncthreads();

    ls[t] = mk ? 1.0f : 0.0f; __syncthreads();
    for (int s = 1; s < S; s <<= 1) {
        float v = ls[t];
        float o = (t + s < S) ? ls[t + s] : 0.0f;
        __syncthreads(); ls[t] = v + o; __syncthreads();
    }
    float sufcnt = ls[t];
    red[t] = mk ? sufcnt : 0.0f; __syncthreads();
    for (int s = 512; s > 0; s >>= 1) { if (t < s) red[t] += red[t + s]; __syncthreads(); }
    float CV = red[0]; __syncthreads();
    float cntMasked = (float)S * (float)S - CV;
    float Mp = (cntMasked > 0.0f) ? fmaxf(Mv, -10.0f) : Mv;

    ls[t] = mk ? __expf(ejv - Mp) : 0.0f; __syncthreads();
    for (int s = 1; s < S; s <<= 1) {
        float v = ls[t];
        float o = (t + s < S) ? ls[t + s] : 0.0f;
        __syncthreads(); ls[t] = v + o; __syncthreads();
    }
    float sufsum = ls[t];
    red[t] = mk ? __expf(eiv - Mp) * sufsum : 0.0f; __syncthreads();
    for (int s = 512; s > 0; s >>= 1) { if (t < s) red[t] += red[t + s]; __syncthreads(); }
    float Zv = red[0];
    if (t == 0) {
        float Z = Zv + cntMasked * __expf(-10.0f - Mp);
        lse_e[b] = Mp + __logf(Z);
    }
}

__global__ __launch_bounds__(256) void out_kernel(
    const float* __restrict__ slp, const float* __restrict__ lse_e,
    const float* __restrict__ e_i, const float* __restrict__ e_j,
    const unsigned char* __restrict__ mask, float* __restrict__ out) {
    const int S = 1024;
    const int bi = blockIdx.x;         // b*S + i
    const int b  = bi >> 10;
    const int i  = bi & (S - 1);
    const float base = slp[bi] + lse_e[b];
    const bool  mi_  = mask[bi] != 0;
    const float eiv  = e_i[bi];
    const unsigned char* mrow = mask + (size_t)b * S;
    const float* ejrow = e_j + (size_t)b * S;
    const int j0 = threadIdx.x * 4;
    float vals[4];
#pragma unroll
    for (int u = 0; u < 4; ++u) {
        int  j     = j0 + u;
        bool valid = mi_ && (mrow[j] != 0) && (j >= i);
        float el   = valid ? (eiv + ejrow[j]) : -10.0f;
        vals[u]    = base - el;
    }
    *(float4*)(out + (size_t)bi * S + j0) = make_float4(vals[0], vals[1], vals[2], vals[3]);
}

extern "C" void kernel_launch(void* const* d_in, const int* in_sizes, int n_in,
                              void* d_out, int out_size, void* d_ws, size_t ws_size,
                              hipStream_t stream) {
    (void)in_sizes; (void)n_in; (void)out_size; (void)ws_size;
    const float*         inputs = (const float*)d_in[0];
    const unsigned char* mask   = (const unsigned char*)d_in[1];  // jnp bool -> 1 byte
    const float*         W0     = (const float*)d_in[2];
    const float*         b0     = (const float*)d_in[3];
    const float*         w1     = (const float*)d_in[4];
    const float*         W2     = (const float*)d_in[5];
    const float*         b2     = (const float*)d_in[6];
    const float*         w3     = (const float*)d_in[7];
    float* out = (float*)d_out;

    const int Bb = 8, S = 1024, D = 1024, U = 1024;
    const int M = Bb * S;   // 8192

    char* p = (char*)d_ws;
    _Float16* A16 = (_Float16*)p;  p += (size_t)M * D * 2;   // inputs, f16
    _Float16* h16 = (_Float16*)p;  p += (size_t)M * U * 2;   // hidden, f16
    _Float16* W0T = (_Float16*)p;  p += (size_t)D * U * 2;   // W0^T, f16
    _Float16* W2T = (_Float16*)p;  p += (size_t)U * U * 2;   // W2^T, f16
    float* sl  = (float*)p;  p += (size_t)M * 4;
    float* ej  = (float*)p;  p += (size_t)M * 4;
    float* ei  = (float*)p;  p += (size_t)M * 4;
    float* slp = (float*)p;  p += (size_t)M * 4;
    float* lse = (float*)p;

    zero_kernel<<<(3 * M + 255) / 256, 256, 0, stream>>>(sl, 3 * M);
    cvt_kernel<<<(M * D / 4 + 255) / 256, 256, 0, stream>>>(inputs, A16, M * D / 4);
    transpose_cvt_kernel<<<dim3(U / 64, D / 64), 256, 0, stream>>>(W0, W0T, D, U);
    transpose_cvt_kernel<<<dim3(U / 64, U / 64), 256, 0, stream>>>(W2, W2T, U, U);

    dim3 gg(U / BN, M / BM);   // (16, 64)
    // h = gelu(X@W0 + b0)*m ; fused: sl = h@w1, ej = h@w3
    fused_gemm_tdm_kernel<0><<<gg, 256, 0, stream>>>(A16, W0T, b0, mask, w1, w3,
                                                     h16, sl, ej, M, U, D);
    // cond = gelu(h@W2 + b2)*m (not stored); fused: ei = cond@w3
    fused_gemm_tdm_kernel<1><<<gg, 256, 0, stream>>>(h16, W2T, b2, mask, nullptr, w3,
                                                     nullptr, nullptr, ei, M, U, U);

    batch_stats_kernel<<<Bb, 1024, 0, stream>>>(sl, ei, ej, mask, slp, lse);
    out_kernel<<<M, 256, 0, stream>>>(slp, lse, ei, ej, mask, out);
}
